// CausalSelfAttention_64020782514808
// MI455X (gfx1250) — matrix operand order repaired and checked by reading, not compiled, not run on hardware
//
#include <hip/hip_runtime.h>

// ---------------------------------------------------------------------------
// CDNA5 (gfx1250) causal self-attention, wave32 WMMA f16->f32 pipeline.
// Staging uses GLOBAL_LOAD_ASYNC_TO_LDS_B128 (ASYNCcnt) when the toolchain
// exposes the builtin; otherwise falls back to sync load+ds_store.
// ---------------------------------------------------------------------------

#define T_SEQ   4096
#define D_MODEL 1024
#define NHEAD   8
#define HDIM    128
#define HTOT    1024          // NHEAD*HDIM
#define ATTN_SCALE 0.12f
#define RMS_EPS 1.1920929e-7f

typedef __attribute__((ext_vector_type(16))) _Float16 v16h;
typedef __attribute__((ext_vector_type(8)))  _Float16 v8h;
typedef __attribute__((ext_vector_type(8)))  float    v8f;

union F16x16 { v16h v; v8h h[2]; };

#if defined(__has_builtin)
#if __has_builtin(__builtin_amdgcn_global_load_async_to_lds_b128) && \
    __has_builtin(__builtin_amdgcn_s_wait_asynccnt)
#define USE_ASYNC_LDS 1
#endif
#endif

// b128 payload type expected by the async builtin: int __vector(4)
typedef int v4i_b128 __attribute__((__vector_size__(16)));
typedef __attribute__((address_space(1))) v4i_b128* g_b128_ptr;
typedef __attribute__((address_space(3))) v4i_b128* l_b128_ptr;

// copy 8 halves (16B) global -> LDS
static __device__ __forceinline__ void cp16_g2l(_Float16* lds_dst,
                                                const _Float16* gsrc) {
#ifdef USE_ASYNC_LDS
    __builtin_amdgcn_global_load_async_to_lds_b128(
        (g_b128_ptr)(void*)gsrc, (l_b128_ptr)(void*)lds_dst, 0, 0);
#else
    *(v8h*)lds_dst = *(const v8h*)gsrc;
#endif
}

static __device__ __forceinline__ void stage_wait() {
#ifdef USE_ASYNC_LDS
    __builtin_amdgcn_s_wait_asynccnt(0);
#endif
}

static __device__ __forceinline__ v8f wmma_f16(v16h a, v16h b, v8f c) {
    return __builtin_amdgcn_wmma_f32_16x16x32_f16(false, a, false, b, (short)0, c,
                                                  false, false);
}

// ---------------------------------------------------------------------------
// K0: f32 -> f16 convert (grid-stride)
// ---------------------------------------------------------------------------
__global__ __launch_bounds__(256) void k_cvt_f16(const float* __restrict__ src,
                                                 _Float16* __restrict__ dst, int n) {
    for (int i = blockIdx.x * 256 + threadIdx.x; i < n; i += gridDim.x * 256)
        dst[i] = (_Float16)src[i];
}

// ---------------------------------------------------------------------------
// K1/K4: C[M,N] (f32) = A[M,K] (f16, row-major) x Bt[N,K]^T (f16, N-major-over-K)
// Block tile 128x64, BK=32, 8 waves; each wave computes a 32x32 patch (2x2 WMMA).
// ---------------------------------------------------------------------------
__global__ __launch_bounds__(256) void k_gemm_f16f32(const _Float16* __restrict__ A,
                                                     const _Float16* __restrict__ Bt,
                                                     float* __restrict__ C,
                                                     int M, int N, int K) {
    __shared__ _Float16 As[128][48];   // 32-wide tiles padded to 48 halves/row
    __shared__ _Float16 Bs[64][48];

    const int tid  = threadIdx.x;
    const int lane = tid & 31;
    const int wid  = tid >> 5;
    const int wm   = wid & 3;          // 4 wave rows  (32 M each)
    const int wn   = wid >> 2;         // 2 wave cols  (32 N each)
    const int bM   = blockIdx.y * 128;
    const int bN   = blockIdx.x * 64;
    const int lh   = lane & 15;
    const bool hiL = lane >= 16;

    v8f acc[2][2] = {};

    for (int k0 = 0; k0 < K; k0 += 32) {
        // stage A block (128x32) : 256 threads x 16 halves
        {
            int r = tid >> 1, cb = (tid & 1) * 16;
            const _Float16* g = A + (size_t)(bM + r) * K + k0 + cb;
            cp16_g2l(&As[r][cb], g);
            cp16_g2l(&As[r][cb + 8], g + 8);
        }
        // stage B block (64x32) : 128 threads x 16 halves
        if (tid < 128) {
            int r = tid >> 1, cb = (tid & 1) * 16;
            const _Float16* g = Bt + (size_t)(bN + r) * K + k0 + cb;
            cp16_g2l(&Bs[r][cb], g);
            cp16_g2l(&Bs[r][cb + 8], g + 8);
        }
        stage_wait();
        __syncthreads();

        // A frag: lanes 0-15 hold K 0-7 & 16-23 of row (lh); lanes 16-31 hold 8-15 & 24-31
        F16x16 af[2], bf[2];
        const int akb = hiL ? 8 : 0;
#pragma unroll
        for (int i = 0; i < 2; ++i) {
            int row = wm * 32 + i * 16 + lh;
            af[i].h[0] = *(const v8h*)&As[row][akb];
            af[i].h[1] = *(const v8h*)&As[row][akb + 16];
        }
        // B frag: lanes 0-15 hold K 0-7 & 16-23 of col (lh); lanes 16-31 hold 8-15 & 24-31
        const int bkb = hiL ? 8 : 0;
#pragma unroll
        for (int j = 0; j < 2; ++j) {
            int col = wn * 32 + j * 16 + lh;
            bf[j].h[0] = *(const v8h*)&Bs[col][bkb];
            bf[j].h[1] = *(const v8h*)&Bs[col][bkb + 16];
        }
#pragma unroll
        for (int i = 0; i < 2; ++i)
#pragma unroll
            for (int j = 0; j < 2; ++j)
                acc[i][j] = wmma_f16(af[i].v, bf[j].v, acc[i][j]);
        __syncthreads();
    }

    // C/D layout: VGPR g -> row g (lanes 0-15) / row 8+g (lanes 16-31), col = lane&15
#pragma unroll
    for (int i = 0; i < 2; ++i) {
        int row0 = bM + wm * 32 + i * 16 + (hiL ? 8 : 0);
#pragma unroll
        for (int j = 0; j < 2; ++j) {
            int col = bN + wn * 32 + j * 16 + lh;
#pragma unroll
            for (int g = 0; g < 8; ++g)
                C[(size_t)(row0 + g) * N + col] = acc[i][j][g];
        }
    }
}

// ---------------------------------------------------------------------------
// K2: per-(t,head) wave: RMS norm q/k, rotary, v = l0*v + l1*ve.
// Writes Qh/Kh as [H][T][HDIM] f16 and V transposed VhT as [H][HDIM][T] f16.
// ---------------------------------------------------------------------------
__global__ __launch_bounds__(256) void k_qkv_post(const float* __restrict__ qkv,
                                                  const float* __restrict__ ve,
                                                  const float* __restrict__ lambdas,
                                                  _Float16* __restrict__ Qh,
                                                  _Float16* __restrict__ Kh,
                                                  _Float16* __restrict__ VhT) {
    const int w    = blockIdx.x * 8 + (threadIdx.x >> 5);
    const int lane = threadIdx.x & 31;
    const int t    = w >> 3;
    const int h    = w & 7;
    const float l0 = lambdas[0], l1 = lambdas[1];

    const float* base = qkv + (size_t)t * (3 * HTOT) + h * HDIM;
    const float* veb  = ve  + (size_t)t * HTOT + h * HDIM;

    const int e0 = lane * 2;                 // handles e0,e0+1 and e0+64,e0+65
    int idx[4] = { e0, e0 + 1, e0 + 64, e0 + 65 };

    float q[4], k[4], vv[4], vex[4];
    float sq = 0.f, sk = 0.f;
#pragma unroll
    for (int i = 0; i < 4; ++i) {
        q[i]   = base[idx[i]];
        k[i]   = base[HTOT + idx[i]];
        vv[i]  = base[2 * HTOT + idx[i]];
        vex[i] = veb[idx[i]];
        sq += q[i] * q[i];
        sk += k[i] * k[i];
    }
#pragma unroll
    for (int m = 16; m >= 1; m >>= 1) {
        sq += __shfl_xor(sq, m);
        sk += __shfl_xor(sk, m);
    }
    const float rq = rsqrtf(sq * (1.0f / HDIM) + RMS_EPS);
    const float rk = rsqrtf(sk * (1.0f / HDIM) + RMS_EPS);
#pragma unroll
    for (int i = 0; i < 4; ++i) { q[i] *= rq; k[i] *= rk; }

    // rotary: pairs (e, e+64); angular[e] = 1024^(-e/31) for e<32, else 0
    float qo[4], ko[4];
#pragma unroll
    for (int p = 0; p < 2; ++p) {
        int e = e0 + p;
        float c = 1.f, s = 0.f;
        if (e < 32) {
            float th = (float)t * powf(1024.0f, -(float)e * (1.0f / 31.0f));
            c = cosf(th);
            s = sinf(th);
        }
        qo[p]     =  q[p] * c + q[2 + p] * s;
        qo[2 + p] = -q[p] * s + q[2 + p] * c;
        ko[p]     =  k[p] * c + k[2 + p] * s;
        ko[2 + p] = -k[p] * s + k[2 + p] * c;
    }

    _Float16* qd = Qh + ((size_t)h * T_SEQ + t) * HDIM;
    _Float16* kd = Kh + ((size_t)h * T_SEQ + t) * HDIM;
#pragma unroll
    for (int i = 0; i < 4; ++i) {
        qd[idx[i]] = (_Float16)qo[i];
        kd[idx[i]] = (_Float16)ko[i];
        float vm = l0 * vv[i] + l1 * vex[i];
        VhT[((size_t)(h * HDIM + idx[i])) * T_SEQ + t] = (_Float16)vm;
    }
}

// ---------------------------------------------------------------------------
// K3: flash attention. Block = (128 q rows x 1 head); 8 waves x 16 q rows each.
// Streams keys in 32-wide tiles: 8 WMMA for S = Q.K^T, online softmax,
// P bounced through LDS (D-layout -> A-layout), 8 WMMA for P.V.
// ---------------------------------------------------------------------------
__global__ __launch_bounds__(256) void k_flash(const _Float16* __restrict__ Qh,
                                               const _Float16* __restrict__ Kh,
                                               const _Float16* __restrict__ VhT,
                                               _Float16* __restrict__ Yh) {
    __shared__ _Float16 Ks[32][144];     // [key][hd]   (B^T of K^T) padded
    __shared__ _Float16 Vts[128][48];    // [hd][key]   (B^T of V)   padded
    __shared__ _Float16 Ps[8][16][48];   // per-wave P scratch [qrow][key]

    const int tid  = threadIdx.x;
    const int lane = tid & 31;
    const int wid  = tid >> 5;
    const int h    = blockIdx.y;
    const int Q0   = blockIdx.x * 128;
    const int q0   = Q0 + wid * 16;
    const int lh   = lane & 15;
    const bool hiL = lane >= 16;
    const int rbase = hiL ? 8 : 0;

    // Q fragments for all four 32-wide k chunks of HDIM, loaded once
    F16x16 qf[4];
    {
        const _Float16* qrow = Qh + ((size_t)h * T_SEQ + q0 + lh) * HDIM;
        const int ab = hiL ? 8 : 0;
#pragma unroll
        for (int c = 0; c < 4; ++c) {
            qf[c].h[0] = *(const v8h*)(qrow + c * 32 + ab);
            qf[c].h[1] = *(const v8h*)(qrow + c * 32 + ab + 16);
        }
    }

    v8f acc[8] = {};
    float mrow[8], lrow[8];
#pragma unroll
    for (int g = 0; g < 8; ++g) { mrow[g] = -3.0e38f; lrow[g] = 0.f; }

    const int kend = Q0 + 128;
    for (int kb = 0; kb < kend; kb += 32) {
        // stage K tile: 32 rows x 128 halves
        {
            int r = tid >> 3, cb = (tid & 7) * 16;
            const _Float16* g = Kh + ((size_t)h * T_SEQ + kb + r) * HDIM + cb;
            cp16_g2l(&Ks[r][cb], g);
            cp16_g2l(&Ks[r][cb + 8], g + 8);
        }
        // stage V^T tile: 128 rows x 32 halves
        {
            int r = tid >> 1, cb = (tid & 1) * 16;
            const _Float16* g = VhT + ((size_t)(h * HDIM + r)) * T_SEQ + kb + cb;
            cp16_g2l(&Vts[r][cb], g);
            cp16_g2l(&Vts[r][cb + 8], g + 8);
        }
        stage_wait();
        __syncthreads();

        if (kb <= q0 + 15) {   // tile has at least one unmasked key for this wave
            // S = Q.K^T : two 16x16 score tiles, reduction over HDIM (4 chunks)
            v8f s[2] = {};
#pragma unroll
            for (int nt = 0; nt < 2; ++nt) {
                int n = nt * 16 + lh;
#pragma unroll
                for (int c = 0; c < 4; ++c) {
                    F16x16 bf;
                    int ko = c * 32 + (hiL ? 8 : 0);
                    bf.h[0] = *(const v8h*)&Ks[n][ko];
                    bf.h[1] = *(const v8h*)&Ks[n][ko + 16];
                    s[nt] = wmma_f16(qf[c].v, bf.v, s[nt]);
                }
            }

            // online softmax (row g lives in lane-half; reduce across 16 lanes)
            const bool full = (kb + 31) <= q0;
            float al[8];
#pragma unroll
            for (int g = 0; g < 8; ++g) {
                float s0 = s[0][g] * ATTN_SCALE;
                float s1 = s[1][g] * ATTN_SCALE;
                if (!full) {
                    int r = q0 + rbase + g;
                    if (kb + lh > r)      s0 = -1.0e30f;
                    if (kb + 16 + lh > r) s1 = -1.0e30f;
                }
                float mx = fmaxf(s0, s1);
#pragma unroll
                for (int msk = 8; msk; msk >>= 1) mx = fmaxf(mx, __shfl_xor(mx, msk));
                float nm = fmaxf(mrow[g], mx);
                float a  = expf(mrow[g] - nm);
                float p0 = expf(s0 - nm);
                float p1 = expf(s1 - nm);
                float rs = p0 + p1;
#pragma unroll
                for (int msk = 8; msk; msk >>= 1) rs += __shfl_xor(rs, msk);
                lrow[g] = lrow[g] * a + rs;
                mrow[g] = nm;
                al[g]   = a;
                Ps[wid][rbase + g][lh]      = (_Float16)p0;
                Ps[wid][rbase + g][16 + lh] = (_Float16)p1;
            }
            asm volatile("s_wait_dscnt 0" ::: "memory");  // P visible within wave

            // rescale running accumulators
#pragma unroll
            for (int j = 0; j < 8; ++j)
#pragma unroll
                for (int g = 0; g < 8; ++g) acc[j][g] *= al[g];

            // P as A-fragment (16 q rows x 32 keys)
            F16x16 pf;
            {
                const int ab = hiL ? 8 : 0;
                pf.h[0] = *(const v8h*)&Ps[wid][lh][ab];
                pf.h[1] = *(const v8h*)&Ps[wid][lh][ab + 16];
            }
            // acc += P.V over the 8 hd chunks
#pragma unroll
            for (int j = 0; j < 8; ++j) {
                F16x16 vf;
                int col = j * 16 + lh;
                int ko  = hiL ? 8 : 0;
                vf.h[0] = *(const v8h*)&Vts[col][ko];
                vf.h[1] = *(const v8h*)&Vts[col][ko + 16];
                acc[j] = wmma_f16(pf.v, vf.v, acc[j]);
            }
        }
        __syncthreads();
    }

    // epilogue: y = acc / l, write f16 [T][HTOT]
#pragma unroll
    for (int j = 0; j < 8; ++j)
#pragma unroll
        for (int g = 0; g < 8; ++g) {
            int q = q0 + rbase + g;
            Yh[(size_t)q * HTOT + h * HDIM + j * 16 + lh] =
                (_Float16)(acc[j][g] / lrow[g]);
        }
}

// ---------------------------------------------------------------------------
// launch
// ---------------------------------------------------------------------------
extern "C" void kernel_launch(void* const* d_in, const int* in_sizes, int n_in,
                              void* d_out, int out_size, void* d_ws, size_t ws_size,
                              hipStream_t stream) {
    const float* x       = (const float*)d_in[0];   // [1,4096,1024]
    const float* ve      = (const float*)d_in[1];   // [1,4096,1024]
    const float* qkv_w   = (const float*)d_in[2];   // [3,1024,1024]
    const float* lambdas = (const float*)d_in[3];   // [2]
    const float* c_proj  = (const float*)d_in[4];   // [1024,1024]
    float* out = (float*)d_out;                     // [1,4096,1024] f32

    // workspace layout (~96 MB)
    char* ws = (char*)d_ws;
    size_t off = 0;
    _Float16* Xh  = (_Float16*)(ws + off); off += (size_t)T_SEQ * D_MODEL * 2;
    _Float16* Wh  = (_Float16*)(ws + off); off += (size_t)3 * HTOT * D_MODEL * 2;
    _Float16* CPW = (_Float16*)(ws + off); off += (size_t)D_MODEL * HTOT * 2;
    float*    QKV = (float*)(ws + off);    off += (size_t)T_SEQ * 3 * HTOT * 4;
    _Float16* Qh  = (_Float16*)(ws + off); off += (size_t)NHEAD * T_SEQ * HDIM * 2;
    _Float16* Kh  = (_Float16*)(ws + off); off += (size_t)NHEAD * T_SEQ * HDIM * 2;
    _Float16* VhT = (_Float16*)(ws + off); off += (size_t)NHEAD * HDIM * T_SEQ * 2;
    _Float16* Yh  = (_Float16*)(ws + off); off += (size_t)T_SEQ * HTOT * 2;
    (void)ws_size; (void)in_sizes; (void)n_in; (void)out_size;

    const int nx = T_SEQ * D_MODEL;
    const int nw = 3 * HTOT * D_MODEL;
    const int nc = D_MODEL * HTOT;
    k_cvt_f16<<<(nx + 255) / 256, 256, 0, stream>>>(x, Xh, nx);
    k_cvt_f16<<<(nw + 255) / 256, 256, 0, stream>>>(qkv_w, Wh, nw);
    k_cvt_f16<<<(nc + 255) / 256, 256, 0, stream>>>(c_proj, CPW, nc);

    // qkv = x . Wqkv^T   (M=4096, N=3072, K=1024)
    k_gemm_f16f32<<<dim3(3 * HTOT / 64, T_SEQ / 128), 256, 0, stream>>>(
        Xh, Wh, QKV, T_SEQ, 3 * HTOT, D_MODEL);

    // rms-norm + rotary + v-mix; one wave per (t, head)
    k_qkv_post<<<(T_SEQ * NHEAD) / 8, 256, 0, stream>>>(QKV, ve, lambdas, Qh, Kh, VhT);

    // flash attention: grid = (q blocks, heads)
    k_flash<<<dim3(T_SEQ / 128, NHEAD), 256, 0, stream>>>(Qh, Kh, VhT, Yh);

    // out = Y . c_proj^T  (M=4096, N=1024, K=1024)
    k_gemm_f16f32<<<dim3(D_MODEL / 64, T_SEQ / 128), 256, 0, stream>>>(
        Yh, CPW, out, T_SEQ, D_MODEL, HTOT);
}
